// BiAffineParser_7078106104314
// MI455X (gfx1250) — matrix operand order repaired and checked by reading, not compiled, not run on hardware
//
#include <hip/hip_runtime.h>
#include <math.h>

// BiAffine span classifier, fused for MI455X (gfx1250, wave32, WMMA).
//
//   kernel 0: pack W2 (768x13 f32) -> f16 pairs [384][16] (cols padded to 16)
//   kernel 1: xs_proj / xe_proj = X @ W1-half, via v_wmma_f32_16x16x4_f32 (full f32)
//   kernel 2: for each (b, i, 16 j's): h = gelu(xs[i]+xe[j]+b1) computed in
//             registers (never hits HBM -> saves ~805MB of traffic), fed as f16
//             A-fragments into v_wmma_f32_16x16x32_f16 against W2, + b2, store.
//
// CDNA5 specifics used: v_wmma_f32_16x16x4_f32, v_wmma_f32_16x16x32_f16,
// global_load_async_to_lds_b128 (+ s_wait_asynccnt) for pure-copy LDS fills.

typedef __attribute__((ext_vector_type(16))) _Float16 v16h;
typedef __attribute__((ext_vector_type(8)))  float    v8f;
typedef __attribute__((ext_vector_type(2)))  float    v2f;

#define Bb 4
#define Ll 256
#define Hh 768
#define NL 13

__device__ __forceinline__ float gelu_exact(float x) {
  // torch-default exact GELU: 0.5*x*(1+erf(x/sqrt(2)))
  return 0.5f * x * (1.0f + erff(x * 0.70710678118654752f));
}

union H2U { unsigned u; _Float16 h[2]; };

// Async global->LDS copy of one 16-byte chunk (ASYNCcnt-tracked, no VGPR staging).
__device__ __forceinline__ void async_copy_b128(unsigned lds_off, const float* gsrc) {
  asm volatile("global_load_async_to_lds_b128 %0, %1, off"
               :: "v"(lds_off), "v"(gsrc)
               : "memory");
}
__device__ __forceinline__ void async_wait0() {
  asm volatile("s_wait_asynccnt 0x0" ::: "memory");
}

// ---------------------------------------------------------------- kernel 0 --
// Pack W2 into [384][16] u32 of f16 pairs: w2p[p][n] = (W2[2p][n], W2[2p+1][n]).
__global__ void pack_w2_kernel(const float* __restrict__ W2,
                               unsigned* __restrict__ w2p) {
  int t = blockIdx.x * blockDim.x + threadIdx.x;
  if (t >= (Hh / 2) * 16) return;
  int p = t >> 4, n = t & 15;
  H2U cv;
  float lo = (n < NL) ? W2[(2 * p) * NL + n] : 0.0f;
  float hi = (n < NL) ? W2[(2 * p + 1) * NL + n] : 0.0f;
  cv.h[0] = (_Float16)lo;
  cv.h[1] = (_Float16)hi;
  w2p[t] = cv.u;
}

// ---------------------------------------------------------------- kernel 1 --
// out(1024x768, per half) = X(1024x768) @ W1half(768x768), f32 WMMA.
// grid (64 Mblocks, 6 col-groups of 128, 2 halves), block 256 = 8 waves.
__global__ void proj_kernel(const float* __restrict__ X,
                            const float* __restrict__ W1,
                            float* __restrict__ out) {
  __shared__ float xt[16 * Hh];  // 48 KB: 16 rows of X, shared by all 8 waves
  const int tid  = threadIdx.x;
  const int lane = tid & 31;
  const int wave = tid >> 5;
  const int mb = blockIdx.x, nb = blockIdx.y, half = blockIdx.z;
  const float* W = W1 + (size_t)half * Hh * Hh;
  float* O = out + (size_t)half * (Bb * Ll) * Hh;

  // Async fill: rows mb*16 .. mb*16+15 of X are contiguous (16*768 floats).
  {
    const float* gsrc = X + (size_t)mb * 16 * Hh;
    const unsigned lds0 = (unsigned)(uintptr_t)(&xt[0]);
    for (int t = tid; t < (16 * Hh) / 4; t += blockDim.x)  // 3072 b128 chunks
      async_copy_b128(lds0 + t * 16, gsrc + t * 4);
    async_wait0();
  }
  __syncthreads();

  // 32-bit A 16x4 layout: lanes 0-15 hold K=0,1; lanes 16-31 hold K=2,3.
  const int mrow = lane & 15;
  const int koff = (lane < 16) ? 0 : 2;
  const int n = nb * 128 + wave * 16 + (lane & 15);

  v8f c = {};
  for (int k0 = 0; k0 < Hh; k0 += 4) {
    v2f a, b;
    a.x = xt[mrow * Hh + k0 + koff];
    a.y = xt[mrow * Hh + k0 + koff + 1];
    b.x = W[(size_t)(k0 + koff) * Hh + n];
    b.y = W[(size_t)(k0 + koff + 1) * Hh + n];
    c = __builtin_amdgcn_wmma_f32_16x16x4_f32(false, a, false, b,
                                              (short)0, c, false, false);
  }
  // 32-bit C/D layout: VGPR r -> row r (lanes 0-15) / r+8 (lanes 16-31).
#pragma unroll
  for (int r = 0; r < 8; ++r) {
    int m = r + ((lane < 16) ? 0 : 8);
    O[(size_t)(mb * 16 + m) * Hh + n] = c[r];
  }
}

// ---------------------------------------------------------------- kernel 2 --
// grid (16 j-blocks, 64 i-blocks, 4 batches), block 128 = 4 waves.
// Wave w handles i = ib*4+w against 16 j's: M=16 pairs, N=16 labels, K=768
// as 24 steps of v_wmma_f32_16x16x32_f16. GELU in fp32, cast to f16 at input.
__global__ void span_kernel(const float* __restrict__ xs_proj,
                            const float* __restrict__ xe_proj,
                            const float* __restrict__ b1,
                            const unsigned* __restrict__ w2p,
                            const float* __restrict__ b2,
                            float* __restrict__ out) {
  __shared__ float xs_s[4][Hh];    // 12 KB: one i-row per wave (async-filled)
  __shared__ float xeb_s[16][Hh];  // 48 KB: 16 j-rows with b1 folded in
  const int tid  = threadIdx.x;
  const int lane = tid & 31;
  const int wave = tid >> 5;
  const int jb = blockIdx.x, ib = blockIdx.y, bb = blockIdx.z;

  // Async fill of xs rows: 4 consecutive rows -> 3072 contiguous floats.
  {
    const float* gsrc = xs_proj + (size_t)(bb * Ll + ib * 4) * Hh;
    const unsigned lds0 = (unsigned)(uintptr_t)(&xs_s[0][0]);
    for (int t = tid; t < (4 * Hh) / 4; t += blockDim.x)  // 768 b128 chunks
      async_copy_b128(lds0 + t * 16, gsrc + t * 4);
  }
  // Synchronous fill of xe rows with b1 folded in (saves a load+add per
  // element in the hot loop; not a pure copy, so not async-able).
  for (int t = tid; t < 16 * Hh; t += blockDim.x) {
    int r = t / Hh, k = t % Hh;
    xeb_s[r][k] = xe_proj[(size_t)(bb * Ll + jb * 16 + r) * Hh + k] + b1[k];
  }
  async_wait0();
  __syncthreads();

  // k order used here for A and B (contiguous): lane<16 -> K = 0..15; lane>=16 -> K = 16..31.
  const int mrow = lane & 15;             // which j-row this lane gathers
  const int klo  = (lane < 16) ? 0 : 16;  // A K-offset (contiguous 16-chunk)
  const int kloB = (lane < 16) ? 0 : 16;  // B K-offset (32x16 layout)
  const int ncol = lane & 15;
  const float* xr = xs_s[wave];
  const float* er = xeb_s[mrow];

  v8f c = {};
  for (int ks = 0; ks < Hh / 32; ++ks) {
    const int k0 = ks * 32;
    v16h a, bm;
#pragma unroll
    for (int t = 0; t < 8; ++t) {
      const int kA = k0 + klo + t;
      a[t]     = (_Float16)gelu_exact(xr[kA]      + er[kA]);
      a[t + 8] = (_Float16)gelu_exact(xr[kA + 8] + er[kA + 8]);
    }
    // B: lane holds W2[k0+kloB .. +15][ncol] as 8 packed f16 pairs (L2-hot).
    const unsigned* wp = &w2p[(size_t)((k0 + kloB) >> 1) * 16 + ncol];
#pragma unroll
    for (int t = 0; t < 8; ++t) {
      H2U cv; cv.u = wp[t * 16];
      bm[2 * t]     = cv.h[0];
      bm[2 * t + 1] = cv.h[1];
    }
    c = __builtin_amdgcn_wmma_f32_16x16x32_f16(false, a, false, bm,
                                               (short)0, c, false, false);
  }

  const int i  = ib * 4 + wave;
  const int j0 = jb * 16;
  if (ncol < NL) {
    const float bias = b2[ncol];
#pragma unroll
    for (int r = 0; r < 8; ++r) {
      int m = r + ((lane < 16) ? 0 : 8);
      out[(((size_t)bb * Ll + i) * Ll + (j0 + m)) * NL + ncol] = c[r] + bias;
    }
  }
}

// ------------------------------------------------------------------ launch --
extern "C" void kernel_launch(void* const* d_in, const int* in_sizes, int n_in,
                              void* d_out, int out_size, void* d_ws, size_t ws_size,
                              hipStream_t stream) {
  const float* X  = (const float*)d_in[0];  // (4,256,768)
  const float* W1 = (const float*)d_in[1];  // (1536,768)
  const float* b1 = (const float*)d_in[2];  // (768)
  const float* W2 = (const float*)d_in[3];  // (768,13)
  const float* b2 = (const float*)d_in[4];  // (13)
  float* out = (float*)d_out;               // (4,256,256,13)

  // Workspace: xs_proj | xe_proj | packed W2  (~6.02 MB total)
  float* xs_proj = (float*)d_ws;
  float* xe_proj = xs_proj + (size_t)(Bb * Ll) * Hh;
  unsigned* w2p  = (unsigned*)(xe_proj + (size_t)(Bb * Ll) * Hh);

  pack_w2_kernel<<<dim3(((Hh / 2) * 16 + 255) / 256), 256, 0, stream>>>(W2, w2p);
  // halves are contiguous in ws: half 0 -> xs_proj, half 1 -> xe_proj
  proj_kernel<<<dim3((Bb * Ll) / 16, Hh / 128, 2), 256, 0, stream>>>(X, W1, xs_proj);
  span_kernel<<<dim3(Ll / 16, Ll / 4, Bb), 128, 0, stream>>>(
      xs_proj, xe_proj, b1, w2p, b2, out);
}
